// MBSFeedForward_82240033784479
// MI455X (gfx1250) — matrix operand order repaired and checked by reading, not compiled, not run on hardware
//
#include <hip/hip_runtime.h>
#include <hip/hip_bf16.h>

typedef __bf16 bf16;
typedef __attribute__((ext_vector_type(16))) __bf16 v16bf;
typedef __attribute__((ext_vector_type(8)))  float  v8f;

#define DM 768
#define DF 3072
#define NB 4
#define NTOK 16384
#define LISTCAP 16448   // 16384 + 4*16 slack for per-expert 16-alignment

// ---------------- bookkeeping kernels ----------------

__global__ void k_init(int* counts, int* cursors, int* list) {
    int i = blockIdx.x * blockDim.x + threadIdx.x;
    if (i < NB) { counts[i] = 0; cursors[i] = 0; }
    if (i < LISTCAP) list[i] = -1;
}

__global__ void k_count(const int* __restrict__ type, int* counts) {
    int i = blockIdx.x * blockDim.x + threadIdx.x;
    if (i < NTOK) {
        int t = type[i];
        if (t > 0) atomicAdd(&counts[t - 1], 1);
    }
}

__global__ void k_offsets(const int* __restrict__ counts, int* offs) {
    int acc = 0;
    for (int e = 0; e < NB; ++e) {
        offs[e] = acc;
        acc += (counts[e] + 15) & ~15;
    }
    offs[NB] = acc;  // total rounded rows
}

__global__ void k_scatter(const int* __restrict__ type, const int* __restrict__ offs,
                          int* cursors, int* list) {
    int i = blockIdx.x * blockDim.x + threadIdx.x;
    if (i < NTOK) {
        int t = type[i];
        if (t > 0) {
            int e = t - 1;
            int p = atomicAdd(&cursors[e], 1);
            list[offs[e] + p] = i;
        }
    }
}

// transpose-convert: per expert z, src is R x C row-major fp32, dst is C x R row-major bf16
__global__ void k_transpose_bf16(const float* __restrict__ src, bf16* __restrict__ dst,
                                 int R, int C) {
    __shared__ float tile[32][33];
    int e = blockIdx.z;
    const float* s = src + (size_t)e * R * C;
    bf16* d = dst + (size_t)e * R * C;
    int c0 = blockIdx.x * 32, r0 = blockIdx.y * 32;
    for (int i = threadIdx.y; i < 32; i += 8) {
        int r = r0 + i, c = c0 + threadIdx.x;
        tile[i][threadIdx.x] = s[(size_t)r * C + c];
    }
    __syncthreads();
    for (int i = threadIdx.y; i < 32; i += 8) {
        int c = c0 + i, r = r0 + threadIdx.x;
        d[(size_t)c * R + r] = (bf16)tile[threadIdx.x][i];
    }
}

// ---------------- GEMM1: h = gelu(x @ W1[e] + b1[e]) over gathered tokens ----------------
// grid: (DF/256, maxTilesM, NB), block 256 (8 waves)
__global__ __launch_bounds__(256) void k_ffn1(
    const float* __restrict__ x, const bf16* __restrict__ w1t,
    const float* __restrict__ b1, const int* __restrict__ counts,
    const int* __restrict__ offs, const int* __restrict__ list,
    bf16* __restrict__ h)
{
    int e = blockIdx.z;
    int tileM = blockIdx.y;
    int cnt = counts[e];
    int rounded = (cnt + 15) & ~15;
    if (tileM * 16 >= rounded) return;
    int rowbase = offs[e] + tileM * 16;  // global gathered row base

    __shared__ bf16 asT[16][DM + 8];     // +8 bf16 (16B) row pad vs bank conflicts
    {
        int r   = threadIdx.x >> 4;
        int seg = threadIdx.x & 15;
        int tok = list[rowbase + r];
        if (tok >= 0) {
            const float* xr = x + (size_t)tok * DM;
            for (int c = seg * 48; c < seg * 48 + 48; ++c) asT[r][c] = (bf16)xr[c];
        } else {
            for (int c = seg * 48; c < seg * 48 + 48; ++c) asT[r][c] = (bf16)0.0f;
        }
    }
    __syncthreads();

    int wid = threadIdx.x >> 5, lane = threadIdx.x & 31;
    int m = lane & 15, hi = lane >> 4;
    int n0 = blockIdx.x * 256 + wid * 32;

    v8f acc0 = {}, acc1 = {};
    const bf16* bbase0 = w1t + ((size_t)e * DF + (n0 + m)) * DM;
    const bf16* bbase1 = bbase0 + (size_t)16 * DM;

    for (int k0 = 0; k0 < DM; k0 += 32) {
        union { v16bf v; uint4 q[2]; } fa, fb0, fb1;
        // A operand: lane holds row m, 16 contiguous K per half-wave (this code's k order)
        fa.q[0] = *(const uint4*)&asT[m][k0 + hi * 16];
        fa.q[1] = *(const uint4*)&asT[m][k0 + hi * 16 + 8];
        // B operand: lane holds column n, 16 contiguous K per half-wave (same k order as A)
        const bf16* bp0 = bbase0 + k0 + hi * 16;
        fb0.q[0] = *(const uint4*)(bp0);
        fb0.q[1] = *(const uint4*)(bp0 + 8);
        const bf16* bp1 = bbase1 + k0 + hi * 16;
        fb1.q[0] = *(const uint4*)(bp1);
        fb1.q[1] = *(const uint4*)(bp1 + 8);
        acc0 = __builtin_amdgcn_wmma_f32_16x16x32_bf16(false, fa.v, false, fb0.v,
                                                       (short)0, acc0, false, false);
        acc1 = __builtin_amdgcn_wmma_f32_16x16x32_bf16(false, fa.v, false, fb1.v,
                                                       (short)0, acc1, false, false);
    }

    int nc0 = n0 + m, nc1 = n0 + 16 + m;
    float bv0 = b1[e * DF + nc0], bv1 = b1[e * DF + nc1];
    for (int j = 0; j < 8; ++j) {
        int row = rowbase + j + hi * 8;    // C/D layout: VGPR j -> M = j + 8*hi
        float v0 = acc0[j] + bv0;
        v0 = 0.5f * v0 * (1.0f + erff(v0 * 0.70710678118654752f));
        h[(size_t)row * DF + nc0] = (bf16)v0;
        float v1 = acc1[j] + bv1;
        v1 = 0.5f * v1 * (1.0f + erff(v1 * 0.70710678118654752f));
        h[(size_t)row * DF + nc1] = (bf16)v1;
    }
}

// ---------------- GEMM2: y = h @ W2[e] + b2; fused expert-LN + outer-LN ----------------
// grid: (LISTCAP/16), block 256 (8 waves, each 6 N-tiles => full 768 columns)
__global__ __launch_bounds__(256) void k_ffn2(
    const float* __restrict__ x, const bf16* __restrict__ h,
    const bf16* __restrict__ w2t, const float* __restrict__ b2,
    const float* __restrict__ ln_g, const float* __restrict__ ln_b,
    const float* __restrict__ out_g, const float* __restrict__ out_b,
    const int* __restrict__ offs, const int* __restrict__ list,
    float* __restrict__ out)
{
    __shared__ bf16 ach[16][256 + 8];
    __shared__ float y[16][DM + 16];

    int rowbase = blockIdx.x * 16;
    if (rowbase >= offs[NB]) return;
    int e = 0;
    for (int i = 1; i < NB; ++i) if (rowbase >= offs[i]) e = i;

    int wid = threadIdx.x >> 5, lane = threadIdx.x & 31;
    int m = lane & 15, hi = lane >> 4;
    int nw = wid * 96;

    v8f zero = {};
    v8f acc[6];
    for (int t = 0; t < 6; ++t) acc[t] = zero;

    for (int kc = 0; kc < DF; kc += 256) {
        __syncthreads();
        {   // stage 16 x 256 bf16 chunk of h
            int linear = threadIdx.x * 16;
            int r = linear >> 8;
            int c = linear & 255;
            const uint4* src = (const uint4*)(h + (size_t)(rowbase + r) * DF + kc + c);
            *(uint4*)&ach[r][c]     = src[0];
            *(uint4*)&ach[r][c + 8] = src[1];
        }
        __syncthreads();
        for (int k0 = 0; k0 < 256; k0 += 32) {
            union { v16bf v; uint4 q[2]; } fa;
            fa.q[0] = *(const uint4*)&ach[m][k0 + hi * 16];
            fa.q[1] = *(const uint4*)&ach[m][k0 + hi * 16 + 8];
            for (int t = 0; t < 6; ++t) {
                int n = nw + t * 16 + m;
                const bf16* bp = w2t + ((size_t)e * DM + n) * DF + kc + k0 + hi * 16;
                union { v16bf v; uint4 q[2]; } fb;
                fb.q[0] = *(const uint4*)bp;
                fb.q[1] = *(const uint4*)(bp + 8);
                acc[t] = __builtin_amdgcn_wmma_f32_16x16x32_bf16(false, fa.v, false, fb.v,
                                                                 (short)0, acc[t], false, false);
            }
        }
    }

    // park GEMM result in LDS for the row-wise LayerNorms
    for (int t = 0; t < 6; ++t) {
        int n = nw + t * 16 + m;
        for (int j = 0; j < 8; ++j) y[j + hi * 8][n] = acc[t][j];
    }
    __syncthreads();

    // 16 threads per row; all 16 live in one half-wave -> shfl width 16
    int r = threadIdx.x >> 4, c16 = threadIdx.x & 15;
    int tok = list[rowbase + r];
    const float* xr = x + (size_t)((tok < 0) ? 0 : tok) * DM;

    // pass 1: t = y + b2 + x ; expert-LN stats
    float s = 0.f, s2 = 0.f;
    for (int c = c16; c < DM; c += 16) {
        float tv = y[r][c] + b2[e * DM + c] + xr[c];
        y[r][c] = tv;
        s += tv; s2 += tv * tv;
    }
    for (int off = 1; off < 16; off <<= 1) {
        s  += __shfl_xor(s,  off, 16);
        s2 += __shfl_xor(s2, off, 16);
    }
    float mean = s / DM;
    float var  = fmaxf(s2 / DM - mean * mean, 0.0f);
    float rs   = rsqrtf(var + 1e-12f);

    // pass 2: sel = LN_e(t); u = sel + x ; outer-LN stats
    float su = 0.f, qu = 0.f;
    for (int c = c16; c < DM; c += 16) {
        float sel = (y[r][c] - mean) * rs * ln_g[e * DM + c] + ln_b[e * DM + c];
        float u = sel + xr[c];
        y[r][c] = u;
        su += u; qu += u * u;
    }
    for (int off = 1; off < 16; off <<= 1) {
        su += __shfl_xor(su, off, 16);
        qu += __shfl_xor(qu, off, 16);
    }
    float m2  = su / DM;
    float v2  = fmaxf(qu / DM - m2 * m2, 0.0f);
    float rs2 = rsqrtf(v2 + 1e-12f);

    if (tok >= 0) {
        for (int c = c16; c < DM; c += 16)
            out[(size_t)tok * DM + c] = (y[r][c] - m2) * rs2 * out_g[c] + out_b[c];
    }
}

// ---------------- type-0 tokens: out = LN_out(x) ----------------
__global__ __launch_bounds__(256) void k_ln_pass(
    const float* __restrict__ x, const int* __restrict__ type,
    const float* __restrict__ out_g, const float* __restrict__ out_b,
    float* __restrict__ out)
{
    int r = threadIdx.x >> 4, c16 = threadIdx.x & 15;
    int tok = blockIdx.x * 16 + r;
    if (tok >= NTOK) return;
    if (type[tok] != 0) return;
    const float* xr = x + (size_t)tok * DM;
    float s = 0.f, s2 = 0.f;
    for (int c = c16; c < DM; c += 16) { float v = xr[c]; s += v; s2 += v * v; }
    for (int off = 1; off < 16; off <<= 1) {
        s  += __shfl_xor(s,  off, 16);
        s2 += __shfl_xor(s2, off, 16);
    }
    float m  = s / DM;
    float v  = fmaxf(s2 / DM - m * m, 0.0f);
    float rs = rsqrtf(v + 1e-12f);
    for (int c = c16; c < DM; c += 16)
        out[(size_t)tok * DM + c] = (xr[c] - m) * rs * out_g[c] + out_b[c];
}

// ---------------- host launch ----------------
extern "C" void kernel_launch(void* const* d_in, const int* in_sizes, int n_in,
                              void* d_out, int out_size, void* d_ws, size_t ws_size,
                              hipStream_t stream) {
    const float* x     = (const float*)d_in[0];
    const int*   type  = (const int*)  d_in[1];
    const float* W1    = (const float*)d_in[2];
    const float* b1    = (const float*)d_in[3];
    const float* W2    = (const float*)d_in[4];
    const float* b2    = (const float*)d_in[5];
    const float* ln_g  = (const float*)d_in[6];
    const float* ln_b  = (const float*)d_in[7];
    const float* out_g = (const float*)d_in[8];
    const float* out_b = (const float*)d_in[9];
    float* out = (float*)d_out;

    char* ws = (char*)d_ws;
    int* counts  = (int*)(ws + 0);
    int* cursors = (int*)(ws + 16);
    int* offs    = (int*)(ws + 32);
    int* list    = (int*)(ws + 64);
    size_t o = 64 + (size_t)LISTCAP * 4;
    o = (o + 255) & ~(size_t)255;
    bf16* w1t = (bf16*)(ws + o); o += (size_t)NB * DF * DM * 2;
    bf16* w2t = (bf16*)(ws + o); o += (size_t)NB * DM * DF * 2;
    bf16* h   = (bf16*)(ws + o); // LISTCAP * DF * 2 bytes

    k_init<<<(LISTCAP + 255) / 256, 256, 0, stream>>>(counts, cursors, list);
    k_count<<<NTOK / 256, 256, 0, stream>>>(type, counts);
    k_offsets<<<1, 1, 0, stream>>>(counts, offs);
    k_scatter<<<NTOK / 256, 256, 0, stream>>>(type, offs, cursors, list);

    // W1[e]: 768x3072 -> w1t[e]: 3072x768 ; W2[e]: 3072x768 -> w2t[e]: 768x3072
    k_transpose_bf16<<<dim3(DF / 32, DM / 32, NB), dim3(32, 8), 0, stream>>>(W1, w1t, DM, DF);
    k_transpose_bf16<<<dim3(DM / 32, DF / 32, NB), dim3(32, 8), 0, stream>>>(W2, w2t, DF, DM);

    k_ffn1<<<dim3(DF / 256, NTOK / 16, NB), 256, 0, stream>>>(x, w1t, b1, counts, offs, list, h);
    k_ffn2<<<LISTCAP / 16, 256, 0, stream>>>(x, h, w2t, b2, ln_g, ln_b, out_g, out_b,
                                             offs, list, out);
    k_ln_pass<<<NTOK / 16, 256, 0, stream>>>(x, type, out_g, out_b, out);
}
